// AttLayer_36421322670272
// MI455X (gfx1250) — matrix operand order repaired and checked by reading, not compiled, not run on hardware
//
#include <hip/hip_runtime.h>
#include <hip/hip_bf16.h>

typedef __attribute__((ext_vector_type(16))) _Float16 v16h;
typedef __attribute__((ext_vector_type(8)))  float    v8f;

#define Bsz   64
#define Tsz   2048
#define Dsz   256
#define Asz   128
#define Msz   (Bsz * Tsz)          // 131072 rows of x
#define WT_PITCH 272               // padded row (halves) to spread LDS banks
#define EPSV  1e-7f

// ---------------------------------------------------------------------------
// Kernel 1: scores.  Each block = 8 waves, each wave does 16 rows (M-tile).
// C = tanh(X W + b) dotted with u, e = exp(.), per-block sum for later norm.
// ---------------------------------------------------------------------------
__global__ __launch_bounds__(256)
void k_scores(const float* __restrict__ X, const float* __restrict__ W,
              const float* __restrict__ Bias, const float* __restrict__ U,
              float* __restrict__ E, float* __restrict__ bsums)
{
    extern __shared__ char smem[];
    _Float16* sWt  = (_Float16*)smem;                       // [128][WT_PITCH]
    float*    sBias = (float*)(smem + Asz * WT_PITCH * 2);  // [128]
    float*    sU    = sBias + Asz;                          // [128]
    float*    sRed  = sU + Asz;                             // [16]

    const int tid = threadIdx.x;

    // Stage W transposed (f16) into LDS: sWt[a][d] = W[d][a]
    for (int idx = tid; idx < Dsz * Asz; idx += 256) {
        int d = idx >> 7;          // 0..255
        int a = idx & (Asz - 1);   // 0..127
        sWt[a * WT_PITCH + d] = (_Float16)W[idx];
    }
    if (tid < Asz) { sBias[tid] = Bias[tid]; sU[tid] = U[tid]; }
    __syncthreads();

    const int wave   = tid >> 5;
    const int lane   = tid & 31;
    const int g      = lane >> 4;       // half-wave group
    const int lanelo = lane & 15;
    const size_t rowbase = (size_t)blockIdx.x * 128 + (size_t)wave * 16;

    v8f acc[8];
#pragma unroll
    for (int nt = 0; nt < 8; ++nt) acc[nt] = (v8f){0.f,0.f,0.f,0.f,0.f,0.f,0.f,0.f};

    const float* xrow = X + (rowbase + (size_t)lanelo) * Dsz;
    const int k0 = g * 16, k1 = g * 16 + 8;

    // K loop: 8 steps of K=32
    for (int kt = 0; kt < 8; ++kt) {
        const float* xr = xrow + kt * 32;
        __builtin_prefetch(xr + 32, 0, 1);   // speculative: global_prefetch_b8

        // A fragment (16x32 f16): lane holds K = g*16+[0,16)
        v16h afr;
#pragma unroll
        for (int i = 0; i < 8; ++i) afr[i]     = (_Float16)xr[k0 + i];
#pragma unroll
        for (int i = 0; i < 8; ++i) afr[8 + i] = (_Float16)xr[k1 + i];

        // B fragments from transposed-W LDS; ds loads for tile nt+1 overlap
        // the matrix-pipe execution of WMMA nt.
#pragma unroll
        for (int nt = 0; nt < 8; ++nt) {
            const _Float16* wt = &sWt[(nt * 16 + lanelo) * WT_PITCH + kt * 32 + g * 16];
            v16h bfr;
#pragma unroll
            for (int i = 0; i < 16; ++i) bfr[i] = wt[i];
            acc[nt] = __builtin_amdgcn_wmma_f32_16x16x32_f16(
                false, afr, false, bfr, (short)0, acc[nt], false, false);
        }
    }

    // Epilogue: p[r] = sum over A of tanh(C + bias)*u, per row m = 8g + r
    // tanh(v) = 1 - 2 * rcp(exp(2v)+1)  -- v_exp_f32 + v_rcp_f32, no IEEE div
    float p[8] = {0.f,0.f,0.f,0.f,0.f,0.f,0.f,0.f};
#pragma unroll
    for (int nt = 0; nt < 8; ++nt) {
        const int n  = nt * 16 + lanelo;
        const float bn = sBias[n];
        const float un = sU[n];
#pragma unroll
        for (int r = 0; r < 8; ++r) {
            float v  = acc[nt][r] + bn;
            float e2 = __expf(2.0f * v);
            float th = 1.0f - 2.0f * __builtin_amdgcn_rcpf(e2 + 1.0f);
            p[r] += th * un;
        }
    }
    // Reduce across the 16 lanes of each half-wave
#pragma unroll
    for (int r = 0; r < 8; ++r) {
#pragma unroll
        for (int off = 1; off < 16; off <<= 1)
            p[r] += __shfl_xor(p[r], off, 32);
    }

    // Lane 0 of each half-wave writes e = exp(score) for rows 8g..8g+7
    if (lanelo == 0) {
        float localsum = 0.f;
        const size_t row = rowbase + 8 * g;
#pragma unroll
        for (int r = 0; r < 8; ++r) {
            float e = __expf(p[r]);
            E[row + r] = e;
            localsum += e;
        }
        sRed[wave * 2 + g] = localsum;
    }
    __syncthreads();
    if (tid == 0) {
        float s = 0.f;
#pragma unroll
        for (int i = 0; i < 16; ++i) s += sRed[i];
        bsums[blockIdx.x] = s;               // 16 blocks per batch, no atomics
    }
}

// ---------------------------------------------------------------------------
// Kernel 2: per-(batch, T-chunk) weighted partial sums over d (coalesced)
// ---------------------------------------------------------------------------
__global__ __launch_bounds__(256)
void k_partial(const float* __restrict__ X, const float* __restrict__ E,
               float* __restrict__ partials)
{
    const int bc = blockIdx.x;      // 0..1023
    const int b  = bc >> 4;
    const int c  = bc & 15;
    const int d  = threadIdx.x;     // 0..255
    const float* xb = X + ((size_t)b * Tsz + (size_t)c * 128) * Dsz;
    const float* eb = E + (size_t)b * Tsz + (size_t)c * 128;
    float acc = 0.f;
#pragma unroll 4
    for (int t = 0; t < 128; ++t)
        acc += xb[(size_t)t * Dsz + d] * eb[t];
    partials[(size_t)bc * Dsz + d] = acc;
}

// ---------------------------------------------------------------------------
// Kernel 3: reduce partials + per-batch sum, apply 1/(sum+eps)
// ---------------------------------------------------------------------------
__global__ __launch_bounds__(256)
void k_final(const float* __restrict__ partials, const float* __restrict__ bsums,
             float* __restrict__ out)
{
    const int b = blockIdx.x;       // 0..63
    const int d = threadIdx.x;      // 0..255
    float s = 0.f;
#pragma unroll
    for (int c = 0; c < 16; ++c) s += partials[(size_t)(b * 16 + c) * Dsz + d];
    float tot = 0.f;
#pragma unroll
    for (int c = 0; c < 16; ++c) tot += bsums[b * 16 + c];
    out[(size_t)b * Dsz + d] = s / (tot + EPSV);
}

// ---------------------------------------------------------------------------
extern "C" void kernel_launch(void* const* d_in, const int* in_sizes, int n_in,
                              void* d_out, int out_size, void* d_ws, size_t ws_size,
                              hipStream_t stream)
{
    const float* x = (const float*)d_in[0];   // [64,2048,256]
    const float* W = (const float*)d_in[1];   // [256,128]
    const float* b = (const float*)d_in[2];   // [128]
    const float* u = (const float*)d_in[3];   // [128,1]
    float* out = (float*)d_out;               // [64,256]

    float* E        = (float*)d_ws;                 // Msz floats
    float* bsums    = E + Msz;                      // 1024 floats
    float* partials = bsums + 1024;                 // 64*16*256 floats
    const size_t need = (size_t)(Msz + 1024 + Bsz * 16 * Dsz) * sizeof(float);
    if (ws_size < need) return;

    const int smemBytes = Asz * WT_PITCH * 2 + (Asz + Asz + 16) * (int)sizeof(float);

    k_scores<<<Msz / 128, 256, smemBytes, stream>>>(x, W, b, u, E, bsums);
    k_partial<<<Bsz * 16, 256, 0, stream>>>(x, E, partials);
    k_final<<<Bsz, 256, 0, stream>>>(partials, bsums, out);
}
